// STMTKF_19567871001297
// MI455X (gfx1250) — matrix operand order repaired and checked by reading, not compiled, not run on hardware
//
#include <hip/hip_runtime.h>
#include <hip/hip_bf16.h>
#include <math.h>

// ---------------------------------------------------------------------------
// MI455X / gfx1250 implementation.
// GEMMs (conv2-as-implicit-GEMM, mamba_proj, in_proj, x_proj, out_proj, m2t)
// run on v_wmma_f32_16x16x32_bf16 (bf16 operands, f32 accumulate).
// Big GEMMs are register-blocked 64Mx32N per wave; fragment loads are grouped
// ahead of the WMMA burst so loads of chunk k+1 overlap WMMAs of chunk k.
// Selective scan parallelized across B*D_INNER=2048 independent recurrences.
// Workspace requirement: ~189 MB (see layout in kernel_launch).
// ---------------------------------------------------------------------------

typedef __bf16 bf16;
typedef __attribute__((ext_vector_type(16))) __bf16 v16bf;
typedef __attribute__((ext_vector_type(8)))  __bf16 v8bf;
typedef __attribute__((ext_vector_type(8)))  float  v8f;

#define BB 4
#define NN 8
#define HH 64
#define WWD 64
#define HO 32
#define WO 32
#define C1 64
#define C2 128
#define LL 8192            // NN*HO*WO tokens per batch
#define NRW 32768          // BB*LL rows for all GEMMs
#define DM 256             // d_model
#define DI 512             // d_inner
#define DST 16             // d_state
#define K2 1728            // 64*27 implicit-GEMM K for conv2

// ---------------------------------------------------------------------------
// Weight packing: f32 -> bf16 row-major [M,K]
// ---------------------------------------------------------------------------
__global__ void k_cvt_bf16(const float* __restrict__ s, bf16* __restrict__ d, int n) {
    int i = blockIdx.x * blockDim.x + threadIdx.x;
    if (i < n) d[i] = (bf16)s[i];
}

// m2t_w (8,256) padded to (16,256) with zero rows
__global__ void k_pack_m2t(const float* __restrict__ s, bf16* __restrict__ d) {
    int i = blockIdx.x * blockDim.x + threadIdx.x;
    if (i < 16 * 256) {
        int r = i >> 8;
        d[i] = (r < 8) ? (bf16)s[i] : (bf16)0.0f;
    }
}

// conv2_w (128,64,3,3,3) -> bf16 [M=128, K=1728], K ordered (kd,kh,kw,ic)
__global__ void k_pack_conv2(const float* __restrict__ w, bf16* __restrict__ d) {
    int i = blockIdx.x * blockDim.x + threadIdx.x;
    if (i < C2 * K2) {
        int m = i / K2, k = i % K2;
        int ic = k & 63, tap = k >> 6;
        int kd = tap / 9, rr = tap % 9, kh = rr / 3, kw = rr % 3;
        d[i] = (bf16)w[(((m * 64 + ic) * 3 + kd) * 3 + kh) * 3 + kw];
    }
}

// ---------------------------------------------------------------------------
// Stage 1: concat(frames, inputs) -> conv3d(4->64, 3x3x3, pad1) -> ReLU ->
// maxpool2x2(H,W). Output feat1 bf16 channel-last [B, N, 32, 32, 64].
// relu(max(a,b)) == max(relu a, relu b), so pool raw conv then ReLU once.
// ---------------------------------------------------------------------------
__global__ void k_conv1(const float* __restrict__ frames, const float* __restrict__ inputs,
                        const float* __restrict__ w, const float* __restrict__ bias,
                        bf16* __restrict__ feat1) {
    int idx = blockIdx.x * blockDim.x + threadIdx.x;
    if (idx >= BB * NN * HO * WO * C1) return;
    int oc = idx % C1;
    int wp = (idx / C1) % WO;
    int hp = (idx / (C1 * WO)) % HO;
    int d  = (idx / (C1 * WO * HO)) % NN;
    int b  = idx / (C1 * WO * HO * NN);
    float best = -1e30f;
    for (int py = 0; py < 2; ++py)
    for (int px = 0; px < 2; ++px) {
        int h = hp * 2 + py, ww = wp * 2 + px;
        float acc = bias[oc];
        for (int kd = 0; kd < 3; ++kd) {
            int dd = d + kd - 1; if (dd < 0 || dd >= NN) continue;
            for (int kh = 0; kh < 3; ++kh) {
                int hh = h + kh - 1; if (hh < 0 || hh >= HH) continue;
                for (int kw = 0; kw < 3; ++kw) {
                    int w2 = ww + kw - 1; if (w2 < 0 || w2 >= WWD) continue;
                    for (int ic = 0; ic < 4; ++ic) {
                        float v = (ic < 3)
                          ? frames[(((size_t)(b * NN + dd) * 3 + ic) * HH + hh) * WWD + w2]
                          : inputs[((size_t)(b * NN + dd) * HH + hh) * WWD + w2];
                        acc += v * w[(((oc * 4 + ic) * 3 + kd) * 3 + kh) * 3 + kw];
                    }
                }
            }
        }
        best = fmaxf(best, acc);
    }
    feat1[idx] = (bf16)fmaxf(best, 0.0f);
}

// ---------------------------------------------------------------------------
// Stage 2: conv2 as implicit GEMM on WMMA. M=128 out-ch, N=32768 pixels,
// K=1728 (27 taps x 64 ic; each 32-wide K chunk = half a tap).
// One wave computes the FULL M column (8 tiles) for one 16-pixel N tile; the
// boundary-checked im2col B-fragment is loaded once per K chunk and reused by
// 8 WMMAs. A-fragments are preloaded in groups of 4 so the loads clause
// together and the WMMA burst overlaps the next group's loads.
// ---------------------------------------------------------------------------
__global__ void k_conv2_wmma(const bf16* __restrict__ Wp, const bf16* __restrict__ feat1,
                             const float* __restrict__ bias, bf16* __restrict__ cnn) {
    int ntile = blockIdx.x * (blockDim.x >> 5) + (threadIdx.x >> 5);
    if (ntile >= NRW / 16) return;
    int lane = threadIdx.x & 31, l = lane & 15, hi = lane >> 4;
    int n = ntile * 16 + l;
    int w0 = n & 31, h0 = (n >> 5) & 31, dd = (n >> 10) & 7, bb = n >> 13;
    const bf16* aB = Wp + (size_t)l * K2 + hi * 16;
    v16bf bzero;
    for (int i = 0; i < 16; ++i) bzero[i] = (bf16)0.0f;
    v8f acc[8];
    for (int mi = 0; mi < 8; ++mi)
        for (int i = 0; i < 8; ++i) acc[mi][i] = 0.0f;
    for (int kc = 0; kc < 54; ++kc) {
        int k = kc * 32;
        int tap = kc >> 1, icoff = (kc & 1) * 32;
        int kd = tap / 9, rr = tap % 9, kh = rr / 3, kw = rr % 3;
        // B fragment (im2col gather, boundary checked) -- loaded once
        int zd = dd + kd - 1, zh = h0 + kh - 1, zw = w0 + kw - 1;
        v16bf bfv = bzero;
        if (zd >= 0 && zd < NN && zh >= 0 && zh < HO && zw >= 0 && zw < WO) {
            const bf16* p = feat1 + (((size_t)(bb * NN + zd) * HO + zh) * WO + zw) * C1
                            + icoff + hi * 16;
            bfv = *(const v16bf*)p;
        }
        // two groups of 4 M tiles: load 4 A-frags, then 4 WMMAs
        #pragma unroll
        for (int g = 0; g < 2; ++g) {
            v16bf af[4];
            #pragma unroll
            for (int mi = 0; mi < 4; ++mi) {
                const bf16* ap = aB + (size_t)(g * 4 + mi) * 16 * K2 + k;
                v8bf a0 = *(const v8bf*)(ap);
                v8bf a1 = *(const v8bf*)(ap + 8);
                for (int i = 0; i < 8; ++i) { af[mi][i] = a0[i]; af[mi][8 + i] = a1[i]; }
            }
            #pragma unroll
            for (int mi = 0; mi < 4; ++mi)
                acc[g * 4 + mi] = __builtin_amdgcn_wmma_f32_16x16x32_bf16(
                    false, af[mi], false, bfv, (short)0, acc[g * 4 + mi], false, false);
        }
    }
    for (int mi = 0; mi < 8; ++mi)
        for (int r = 0; r < 8; ++r) {
            int m = mi * 16 + r + (hi ? 8 : 0);
            float v = fmaxf(acc[mi][r] + bias[m], 0.0f);
            cnn[(size_t)n * C2 + m] = (bf16)v;
        }
}

// ---------------------------------------------------------------------------
// Register-blocked bf16 WMMA GEMM: 64(M) x 32(N) per wave (requires M%64==0).
// Y[n, m] = act[n, :K] . W[m, :K] (+bias).  Per K chunk: load 2 B-frags + 4
// A-frags (one clause, one wait), then 8 back-to-back WMMAs; next chunk's
// loads issue while these WMMAs drain.
// A-frag: lane(l,hi): row M=l, K = hi*16..hi*16+15.
// B-frag: lane(l,hi): col N=l, K = hi*16..hi*16+15 (contiguous 32B load).
// ---------------------------------------------------------------------------
__global__ void k_gemm_blk(const bf16* __restrict__ W, const bf16* __restrict__ X,
                           bf16* __restrict__ Y, const float* __restrict__ bias,
                           int M, int N, int K) {
    int wave = blockIdx.x * (blockDim.x >> 5) + (threadIdx.x >> 5);
    int MB = M >> 6, NB = N >> 5;
    if (wave >= MB * NB) return;
    int mblk = wave % MB, nblk = wave / MB;
    int lane = threadIdx.x & 31, l = lane & 15, hi = lane >> 4;
    const bf16* aB = W + (size_t)(mblk * 64 + l) * K + hi * 16;
    const bf16* bB0 = X + (size_t)(nblk * 32 + l) * K + hi * 16;
    const bf16* bB1 = bB0 + (size_t)16 * K;
    v8f acc[8];
    for (int i = 0; i < 8; ++i)
        for (int j = 0; j < 8; ++j) acc[i][j] = 0.0f;
    for (int k = 0; k < K; k += 32) {
        v16bf b0 = *(const v16bf*)(bB0 + k);
        v16bf b1 = *(const v16bf*)(bB1 + k);
        v16bf af[4];
        #pragma unroll
        for (int mi = 0; mi < 4; ++mi) {
            const bf16* ap = aB + (size_t)mi * 16 * K + k;
            v8bf a0 = *(const v8bf*)(ap);
            v8bf a1 = *(const v8bf*)(ap + 8);
            for (int i = 0; i < 8; ++i) { af[mi][i] = a0[i]; af[mi][8 + i] = a1[i]; }
        }
        #pragma unroll
        for (int mi = 0; mi < 4; ++mi) {
            acc[mi * 2 + 0] = __builtin_amdgcn_wmma_f32_16x16x32_bf16(
                false, af[mi], false, b0, (short)0, acc[mi * 2 + 0], false, false);
            acc[mi * 2 + 1] = __builtin_amdgcn_wmma_f32_16x16x32_bf16(
                false, af[mi], false, b1, (short)0, acc[mi * 2 + 1], false, false);
        }
    }
    for (int mi = 0; mi < 4; ++mi)
        for (int ni = 0; ni < 2; ++ni) {
            int n = nblk * 32 + ni * 16 + l;
            for (int r = 0; r < 8; ++r) {
                int m = mblk * 64 + mi * 16 + r + (hi ? 8 : 0);
                float v = acc[mi * 2 + ni][r] + (bias ? bias[m] : 0.0f);
                Y[(size_t)n * M + m] = (bf16)v;
            }
        }
}

// ---------------------------------------------------------------------------
// Simple bf16 WMMA GEMM (one 16x16 tile per wave) for small-M layers
// (x_proj M=48, m2t M=16-padded).
// ---------------------------------------------------------------------------
__global__ void k_gemm(const bf16* __restrict__ W, const bf16* __restrict__ X,
                       void* __restrict__ Y, const float* __restrict__ bias,
                       int M, int N, int K, int ldY, int Mstore, int storeF32) {
    int wave = blockIdx.x * (blockDim.x >> 5) + (threadIdx.x >> 5);
    int MT = M >> 4, NT = N >> 4;
    if (wave >= MT * NT) return;
    int mtile = wave % MT, ntile = wave / MT;
    int lane = threadIdx.x & 31, l = lane & 15, hi = lane >> 4;
    const bf16* aptr = W + (size_t)(mtile * 16 + l) * K + hi * 16;
    const bf16* bptr = X + (size_t)(ntile * 16 + l) * K + hi * 16;
    v8f acc; for (int i = 0; i < 8; ++i) acc[i] = 0.0f;
    for (int k = 0; k < K; k += 32) {
        __builtin_prefetch(bptr + k + 128, 0, 0);
        v8bf a0 = *(const v8bf*)(aptr + k);
        v8bf a1 = *(const v8bf*)(aptr + k + 8);
        v16bf af; for (int i = 0; i < 8; ++i) { af[i] = a0[i]; af[8 + i] = a1[i]; }
        v16bf bf_ = *(const v16bf*)(bptr + k);
        acc = __builtin_amdgcn_wmma_f32_16x16x32_bf16(false, af, false, bf_,
                                                      (short)0, acc, false, false);
    }
    int n = ntile * 16 + l;
    for (int r = 0; r < 8; ++r) {
        int m = mtile * 16 + r + (hi ? 8 : 0);
        if (m < Mstore) {
            float v = acc[r] + (bias ? bias[m] : 0.0f);
            if (storeF32) ((float*)Y)[(size_t)n * ldY + m] = v;
            else          ((bf16*)Y)[(size_t)n * ldY + m] = (bf16)v;
        }
    }
}

// ---------------------------------------------------------------------------
// Depthwise causal conv1d (k=4) over sequence + SiLU.  xi = xz cols [0,512).
// ---------------------------------------------------------------------------
__global__ void k_dwconv(const bf16* __restrict__ xz, const float* __restrict__ w,
                         const float* __restrict__ bias, bf16* __restrict__ xc) {
    int idx = blockIdx.x * blockDim.x + threadIdx.x;
    if (idx >= BB * LL * DI) return;
    int d = idx % DI;
    int t = (idx / DI) % LL;
    int b = idx / (DI * LL);
    float acc = bias[d];
    for (int j = 0; j < 4; ++j) {
        int tt = t - 3 + j;
        if (tt >= 0)
            acc += w[d * 4 + j] * (float)xz[(size_t)(b * LL + tt) * 1024 + d];
    }
    acc = acc / (1.0f + __expf(-acc));   // SiLU
    xc[(size_t)(b * LL + t) * DI + d] = (bf16)acc;
}

// ---------------------------------------------------------------------------
// Selective scan. 2048 independent recurrences (b,d), h[16] in registers.
// Fuses dt_proj+softplus, the scan, skip (u*D), and SiLU(z) gating.
// dbc rows (48 f32, shared by all d of a batch) staged through LDS in chunks.
// ---------------------------------------------------------------------------
__global__ void k_scan(const float* __restrict__ dbc, const bf16* __restrict__ xc,
                       const bf16* __restrict__ xz, const float* __restrict__ dtw,
                       const float* __restrict__ dtb, const float* __restrict__ A_log,
                       const float* __restrict__ Dsk, bf16* __restrict__ yg) {
    int b = blockIdx.x >> 1;
    int d = ((blockIdx.x & 1) * 256) + threadIdx.x;
    float A[DST], Wdt[DST], h[DST];
    for (int s = 0; s < DST; ++s) {
        A[s] = -__expf(A_log[d * DST + s]);
        Wdt[s] = dtw[d * DST + s];
        h[s] = 0.0f;
    }
    float bdt = dtb[d], Dd = Dsk[d];
    __shared__ float sdbc[32 * 48];
    for (int t0 = 0; t0 < LL; t0 += 32) {
        __syncthreads();
        const float* src = dbc + (size_t)(b * LL + t0) * 48;
        for (int i = threadIdx.x; i < 32 * 48; i += 256) sdbc[i] = src[i];
        __syncthreads();
        for (int tt = 0; tt < 32; ++tt) {
            const float* row = &sdbc[tt * 48];
            float dt = bdt;
            #pragma unroll
            for (int s = 0; s < DST; ++s) dt += row[s] * Wdt[s];
            dt = (dt > 20.0f) ? dt : log1pf(__expf(dt));   // softplus
            size_t n = (size_t)b * LL + (t0 + tt);
            float u = (float)xc[n * DI + d];
            float du = dt * u;
            float y = 0.0f;
            #pragma unroll
            for (int s = 0; s < DST; ++s) {
                float dA = __expf(dt * A[s]);
                h[s] = dA * h[s] + du * row[16 + s];
                y += h[s] * row[32 + s];
            }
            float z = (float)xz[n * 1024 + DI + d];
            float g = z / (1.0f + __expf(-z));
            yg[n * DI + d] = (bf16)((y + u * Dd) * g);
        }
    }
}

// ---------------------------------------------------------------------------
// Bounding-box measurements from masks inputs(B,N,64,64) > 0.5.
// ---------------------------------------------------------------------------
__global__ void k_meas(const float* __restrict__ inputs, float* __restrict__ meas) {
    int b = blockIdx.x / NN, nf = blockIdx.x % NN;
    __shared__ int rmin, rmax, cmin, cmax, anyv;
    if (threadIdx.x == 0) { rmin = 64; rmax = -1; cmin = 64; cmax = -1; anyv = 0; }
    __syncthreads();
    const float* base = inputs + ((size_t)(b * NN + nf) * HH) * WWD;
    int t = threadIdx.x;   // 64 threads: row t and col t
    int ra = 0, ca = 0;
    for (int i = 0; i < 64; ++i) {
        if (base[t * WWD + i] > 0.5f) ra = 1;
        if (base[i * WWD + t] > 0.5f) ca = 1;
    }
    if (ra) { atomicMin(&rmin, t); atomicMax(&rmax, t); atomicExch(&anyv, 1); }
    if (ca) { atomicMin(&cmin, t); atomicMax(&cmax, t); }
    __syncthreads();
    if (threadIdx.x == 0) {
        float* m = meas + (b * NN + nf) * 4;
        if (anyv) {
            float xmin = (float)cmin, xmax = (float)cmax;
            float ymin = (float)rmin, ymax = (float)rmax;
            m[0] = (xmin + xmax) * 0.5f / 64.0f;
            m[1] = (ymin + ymax) * 0.5f / 64.0f;
            m[2] = (xmax - xmin + 1.0f) / 64.0f;
            m[3] = (ymax - ymin + 1.0f) / 64.0f;
        } else {
            m[0] = m[1] = m[2] = m[3] = 0.0f;
        }
    }
}

// ---------------------------------------------------------------------------
// Cross-attention (online softmax over L=8192) + delta_F + Kalman scan.
// One block (256 threads) per batch.
// ---------------------------------------------------------------------------
__global__ void k_attn_kf(const float* __restrict__ xtok,
                          const float* __restrict__ qw, const float* __restrict__ qb,
                          const float* __restrict__ kw, const float* __restrict__ kb,
                          const float* __restrict__ vw, const float* __restrict__ vb,
                          const float* __restrict__ lt,
                          const float* __restrict__ kfw, const float* __restrict__ kfb,
                          const float* __restrict__ meas, float* __restrict__ state_out) {
    int b = blockIdx.x;
    __shared__ float q[64], mot[64], Fms[64];
    __shared__ float red[256 * 10];
    if (threadIdx.x < 64) {
        int t = threadIdx.x / 8, j = threadIdx.x % 8;
        float a = qb[j];
        for (int i = 0; i < 8; ++i) a += lt[t * 8 + i] * qw[j * 8 + i];
        q[t * 8 + j] = a;
    }
    __syncthreads();
    const float scale = 0.35355339059327373f; // 1/sqrt(8)
    float mmax[8], ssum[8], vacc[8][8];
    for (int t = 0; t < 8; ++t) { mmax[t] = -1e30f; ssum[t] = 0.0f;
                                  for (int j = 0; j < 8; ++j) vacc[t][j] = 0.0f; }
    for (int l = threadIdx.x; l < LL; l += 256) {
        const float* xr = xtok + ((size_t)b * LL + l) * 8;
        float kl[8], vl[8];
        for (int j = 0; j < 8; ++j) {
            float ka = kb[j], va = vb[j];
            for (int i = 0; i < 8; ++i) { ka += xr[i] * kw[j * 8 + i]; va += xr[i] * vw[j * 8 + i]; }
            kl[j] = ka; vl[j] = va;
        }
        for (int t = 0; t < 8; ++t) {
            float s = 0.0f;
            for (int j = 0; j < 8; ++j) s += q[t * 8 + j] * kl[j];
            s *= scale;
            float nm = fmaxf(mmax[t], s);
            float corr = __expf(mmax[t] - nm), e = __expf(s - nm);
            ssum[t] = ssum[t] * corr + e;
            for (int j = 0; j < 8; ++j) vacc[t][j] = vacc[t][j] * corr + e * vl[j];
            mmax[t] = nm;
        }
    }
    for (int t = 0; t < 8; ++t) {
        float* r = &red[threadIdx.x * 10];
        r[0] = mmax[t]; r[1] = ssum[t];
        for (int j = 0; j < 8; ++j) r[2 + j] = vacc[t][j];
        __syncthreads();
        for (int off = 128; off > 0; off >>= 1) {
            if (threadIdx.x < off) {
                float* a = &red[threadIdx.x * 10];
                float* c = &red[(threadIdx.x + off) * 10];
                float nm = fmaxf(a[0], c[0]);
                float ea = __expf(a[0] - nm), ec = __expf(c[0] - nm);
                a[1] = a[1] * ea + c[1] * ec;
                for (int j = 0; j < 8; ++j) a[2 + j] = a[2 + j] * ea + c[2 + j] * ec;
                a[0] = nm;
            }
            __syncthreads();
        }
        if (threadIdx.x < 8) mot[t * 8 + threadIdx.x] = red[2 + threadIdx.x] / red[1];
        __syncthreads();
    }
    if (threadIdx.x < 64) {
        int p = threadIdx.x;
        float dv = kfb[p];
        for (int k = 0; k < 8; ++k) dv += mot[k] * kfw[p * 8 + k];  // motion token 0
        Fms[p] = dv + ((p / 8 == p % 8) ? 1.0f : 0.0f);
    }
    __syncthreads();
    if (threadIdx.x == 0) {
        float st[8]; for (int i = 0; i < 8; ++i) st[i] = 0.0f;
        for (int nf = 0; nf < NN; ++nf) {
            float ns[8];
            for (int i = 0; i < 8; ++i) {
                float a = 0.0f;
                for (int j = 0; j < 8; ++j) a += Fms[i * 8 + j] * st[j];
                ns[i] = a;
            }
            for (int i = 0; i < 8; ++i) st[i] = ns[i];
            for (int k = 0; k < 4; ++k)
                st[k] += 0.1f * (meas[(b * NN + nf) * 4 + k] - ns[k]);
        }
        for (int i = 0; i < 8; ++i) state_out[b * 8 + i] = st[i];
    }
}

// ---------------------------------------------------------------------------
// Refinement: concat(cnn last frame, state broadcast) -> conv3x3(136->64)+ReLU
// -> conv1x1(64->1) -> sigmoid. One thread per output pixel, both convs fused.
// ---------------------------------------------------------------------------
__global__ void k_refine(const bf16* __restrict__ cnn, const float* __restrict__ state,
                         const float* __restrict__ w1, const float* __restrict__ b1,
                         const float* __restrict__ w2, const float* __restrict__ b2,
                         float* __restrict__ out) {
    int idx = blockIdx.x * blockDim.x + threadIdx.x;
    if (idx >= BB * HO * WO) return;
    int w0 = idx % WO, h0 = (idx / WO) % HO, b = idx / (HO * WO);
    float oacc = b2[0];
    for (int oc = 0; oc < 64; ++oc) {
        float a = b1[oc];
        for (int kh = 0; kh < 3; ++kh) {
            int hh = h0 + kh - 1; if (hh < 0 || hh >= HO) continue;
            for (int kwp = 0; kwp < 3; ++kwp) {
                int ww = w0 + kwp - 1; if (ww < 0 || ww >= WO) continue;
                const bf16* pix = cnn + (((size_t)(b * NN + (NN - 1)) * HO + hh) * WO + ww) * C2;
                const float* wrow = w1 + ((size_t)oc * 136 * 9) + kh * 3 + kwp;
                for (int ic = 0; ic < C2; ++ic)
                    a += (float)pix[ic] * wrow[ic * 9];
                for (int ic = 0; ic < 8; ++ic)
                    a += state[b * 8 + ic] * wrow[(C2 + ic) * 9];
            }
        }
        oacc += fmaxf(a, 0.0f) * w2[oc];
    }
    out[idx] = 1.0f / (1.0f + __expf(-oacc));
}

// ---------------------------------------------------------------------------
extern "C" void kernel_launch(void* const* d_in, const int* in_sizes, int n_in,
                              void* d_out, int out_size, void* d_ws, size_t ws_size,
                              hipStream_t stream) {
    const float* frames   = (const float*)d_in[0];
    const float* inputs   = (const float*)d_in[1];
    const float* conv1_w  = (const float*)d_in[2];
    const float* conv1_b  = (const float*)d_in[3];
    const float* conv2_w  = (const float*)d_in[4];
    const float* conv2_b  = (const float*)d_in[5];
    const float* mproj_w  = (const float*)d_in[6];
    const float* mproj_b  = (const float*)d_in[7];
    const float* inproj_w = (const float*)d_in[8];
    const float* c1d_w    = (const float*)d_in[9];
    const float* c1d_b    = (const float*)d_in[10];
    const float* xproj_w  = (const float*)d_in[11];
    const float* dtproj_w = (const float*)d_in[12];
    const float* dtproj_b = (const float*)d_in[13];
    const float* A_log    = (const float*)d_in[14];
    const float* Dsk      = (const float*)d_in[15];
    const float* oproj_w  = (const float*)d_in[16];
    const float* m2t_w    = (const float*)d_in[17];
    const float* m2t_b    = (const float*)d_in[18];
    const float* q_w      = (const float*)d_in[19];
    const float* q_b      = (const float*)d_in[20];
    const float* k_w      = (const float*)d_in[21];
    const float* k_b      = (const float*)d_in[22];
    const float* v_w      = (const float*)d_in[23];
    const float* v_b      = (const float*)d_in[24];
    const float* lt       = (const float*)d_in[25];
    const float* kf_w     = (const float*)d_in[26];
    const float* kf_b     = (const float*)d_in[27];
    const float* ref1_w   = (const float*)d_in[28];
    const float* ref1_b   = (const float*)d_in[29];
    const float* ref2_w   = (const float*)d_in[30];
    const float* ref2_b   = (const float*)d_in[31];
    float* out = (float*)d_out;

    // --- workspace layout (~189 MB, 256B aligned) ---
    char* ws = (char*)d_ws;
    size_t off = 0;
    auto alloc = [&](size_t bytes) { size_t r = off; off = (off + bytes + 255) & ~(size_t)255; return r; };
    bf16* w_conv2  = (bf16*)(ws + alloc((size_t)C2 * K2 * 2));
    bf16* w_mamba  = (bf16*)(ws + alloc((size_t)DM * C2 * 2));
    bf16* w_inproj = (bf16*)(ws + alloc((size_t)1024 * DM * 2));
    bf16* w_xproj  = (bf16*)(ws + alloc((size_t)48 * DI * 2));
    bf16* w_oproj  = (bf16*)(ws + alloc((size_t)DM * DI * 2));
    bf16* w_m2t    = (bf16*)(ws + alloc((size_t)16 * DM * 2));
    bf16* feat1    = (bf16*)(ws + alloc((size_t)BB * NN * HO * WO * C1 * 2));
    bf16* cnn      = (bf16*)(ws + alloc((size_t)NRW * C2 * 2));
    bf16* xm       = (bf16*)(ws + alloc((size_t)NRW * DM * 2));
    bf16* xz       = (bf16*)(ws + alloc((size_t)NRW * 1024 * 2));
    bf16* xc       = (bf16*)(ws + alloc((size_t)NRW * DI * 2));
    float* dbc     = (float*)(ws + alloc((size_t)NRW * 48 * 4));
    bf16* yg       = (bf16*)(ws + alloc((size_t)NRW * DI * 2));
    bf16* outm     = (bf16*)(ws + alloc((size_t)NRW * DM * 2));
    float* xtok    = (float*)(ws + alloc((size_t)NRW * 8 * 4));
    float* meas    = (float*)(ws + alloc((size_t)BB * NN * 4 * 4));
    float* state   = (float*)(ws + alloc((size_t)BB * 8 * 4));

    // --- weight packing ---
    k_pack_conv2<<<(C2 * K2 + 255) / 256, 256, 0, stream>>>(conv2_w, w_conv2);
    k_cvt_bf16<<<(DM * C2 + 255) / 256, 256, 0, stream>>>(mproj_w, w_mamba, DM * C2);
    k_cvt_bf16<<<(1024 * DM + 255) / 256, 256, 0, stream>>>(inproj_w, w_inproj, 1024 * DM);
    k_cvt_bf16<<<(48 * DI + 255) / 256, 256, 0, stream>>>(xproj_w, w_xproj, 48 * DI);
    k_cvt_bf16<<<(DM * DI + 255) / 256, 256, 0, stream>>>(oproj_w, w_oproj, DM * DI);
    k_pack_m2t<<<(16 * DM + 255) / 256, 256, 0, stream>>>(m2t_w, w_m2t);

    // --- stage 1: conv1 + relu + maxpool ---
    k_conv1<<<(BB * NN * HO * WO * C1) / 256, 256, 0, stream>>>(frames, inputs, conv1_w, conv1_b, feat1);

    // --- stage 2: conv2 implicit GEMM (WMMA, full-M per wave) ---
    k_conv2_wmma<<<(NRW / 16) / 8, 256, 0, stream>>>(w_conv2, feat1, conv2_b, cnn);

    // --- mamba_proj: [32768,128] x [256,128]^T + b -> xm bf16 (blocked) ---
    k_gemm_blk<<<((DM / 64) * (NRW / 32)) / 8, 256, 0, stream>>>(w_mamba, cnn, xm, mproj_b,
                                                                 DM, NRW, C2);
    // --- in_proj: [32768,256] x [1024,256]^T -> xz bf16 (blocked) ---
    k_gemm_blk<<<((1024 / 64) * (NRW / 32)) / 8, 256, 0, stream>>>(w_inproj, xm, xz, nullptr,
                                                                   1024, NRW, DM);
    // --- depthwise causal conv + SiLU -> xc ---
    k_dwconv<<<(BB * LL * DI) / 256, 256, 0, stream>>>(xz, c1d_w, c1d_b, xc);
    // --- x_proj: [32768,512] x [48,512]^T -> dbc f32 ---
    k_gemm<<<((48 / 16) * (NRW / 16)) / 8, 256, 0, stream>>>(w_xproj, xc, dbc, nullptr,
                                                             48, NRW, DI, 48, 48, 1);
    // --- selective scan (fused dt_proj/softplus/skip/gate) -> yg bf16 ---
    k_scan<<<8, 256, 0, stream>>>(dbc, xc, xz, dtproj_w, dtproj_b, A_log, Dsk, yg);
    // --- out_proj: [32768,512] x [256,512]^T -> outm bf16 (blocked) ---
    k_gemm_blk<<<((DM / 64) * (NRW / 32)) / 8, 256, 0, stream>>>(w_oproj, yg, outm, nullptr,
                                                                 DM, NRW, DI);
    // --- m2t: [32768,256] x [16(pad),256]^T -> xtok f32 (store 8 cols) ---
    k_gemm<<<((16 / 16) * (NRW / 16)) / 8, 256, 0, stream>>>(w_m2t, outm, xtok, m2t_b,
                                                             16, NRW, DM, 8, 8, 1);
    // --- measurements, attention + Kalman, refine ---
    k_meas<<<BB * NN, 64, 0, stream>>>(inputs, meas);
    k_attn_kf<<<BB, 256, 0, stream>>>(xtok, q_w, q_b, k_w, k_b, v_w, v_b, lt,
                                      kf_w, kf_b, meas, state);
    k_refine<<<(BB * HO * WO + 127) / 128, 128, 0, stream>>>(cnn, state, ref1_w, ref1_b,
                                                             ref2_w, ref2_b, out);
    (void)in_sizes; (void)n_in; (void)out_size; (void)ws_size;
}
